// AdjacencyGenerator_77206332113064
// MI455X (gfx1250) — compile-verified, matrix operand order checked by reading, not run on hardware
//
#include <hip/hip_runtime.h>

// ---------------------------------------------------------------------------
// AdjacencyGenerator for MI455X (gfx1250, wave32, WMMA)
//   B = 8192 nodes, D = 1024 features, top-(K+1)=21 neighbors.
//
// Pipeline:
//   1) normalize_rows : f32 rows -> L2-normalized f16 rows (16 MB, L2-resident)
//   2) gemm_topk      : fused  xn @ xn^T (WMMA f16->f32)  +  per-row top-21
//   3) zero_out       : clear 256 MB output
//   4) scatter_w      : out[row, idx[r]] = ew[r] / (sum(ew)+1e-8)
// ---------------------------------------------------------------------------

typedef __attribute__((ext_vector_type(16))) _Float16 v16h;
typedef __attribute__((ext_vector_type(8)))  float    v8f;

#define BN   8192   // nodes
#define DK   1024   // feature dim
#define KSEL 21     // K+1 neighbors kept per row
#define IDX_PITCH 32  // padded ints per row for the index table

// ---------------- Kernel 1: row L2-normalize, f32 -> f16 -------------------
__global__ void normalize_rows(const float* __restrict__ x,
                               _Float16* __restrict__ xn) {
  __shared__ float red[256];
  const int row = blockIdx.x;
  const int tid = threadIdx.x;
  const float* xr = x + (size_t)row * DK;

  float s = 0.f;
  for (int c = tid; c < DK; c += 256) { float v = xr[c]; s += v * v; }
  red[tid] = s;
  __syncthreads();
  for (int off = 128; off > 0; off >>= 1) {
    if (tid < off) red[tid] += red[tid + off];
    __syncthreads();
  }
  const float inv = 1.f / fmaxf(sqrtf(red[0]), 1e-12f);  // matches reference eps
  _Float16* outr = xn + (size_t)row * DK;
  for (int c = tid; c < DK; c += 256) outr[c] = (_Float16)(xr[c] * inv);
}

// ---------------- Kernel 2: fused Gram-GEMM + per-row top-21 ---------------
// One workgroup (4 waves, 128 threads) owns a 16-row M-tile.
// Loop over N in chunks of 64 columns; wave w computes the 16x16 tile at
// columns nc + 16*w via 32 chained V_WMMA_F32_16X16X32_F16 ops (K=1024).
// A slab (16 x 1024 f16) is staged in LDS once; B rows stream from L2.
// Gram symmetry => A and B fragments use the identical per-lane load pattern,
// so any K-permutation cancels and the dot products are exact.
__launch_bounds__(128)
__global__ void gemm_topk(const _Float16* __restrict__ xn,
                          int* __restrict__ topk_idx) {
  __shared__ __align__(32) _Float16 Alds[16 * DK];   // 32 KB
  __shared__ float tile[16 * 64];                    // 4 KB: 16 rows x 64 cols
  __shared__ float tv[16 * KSEL];                    // per-row top values (desc)
  __shared__ int   ti[16 * KSEL];                    // per-row top indices

  const int tid  = threadIdx.x;      // 0..127
  const int wave = tid >> 5;         // 0..3
  const int lane = tid & 31;
  const int mt   = blockIdx.x;       // M tile id
  const int row0 = mt * 16;

  // Stage A slab: 16 rows x 1024 halves = 2048 uint4
  {
    const uint4* src = (const uint4*)(xn + (size_t)row0 * DK);
    uint4*       dst = (uint4*)Alds;
    for (int i = tid; i < 2048; i += 128) dst[i] = src[i];
  }
  if (tid < 16 * KSEL) { tv[tid] = -3.0e38f; ti[tid] = 0; }
  __syncthreads();

  // Fragment addressing (identical pattern for A and B):
  //   lane L supplies matrix row (L & 15), K-halves [koff, koff+16) per step.
  const int m    = lane & 15;
  const int koff = (lane >> 4) * 16;
  const _Float16* aptr = Alds + m * DK + koff;

  for (int nc = 0; nc < BN; nc += 64) {
    const int colrow = nc + wave * 16 + m;   // global column this lane feeds B
    const _Float16* bptr = xn + (size_t)colrow * DK + koff;

    v8f acc = {};
    for (int ks = 0; ks < DK; ks += 32) {
      v16h a = *(const v16h*)(aptr + ks);    // ds_load x2 (b128)
      v16h b = *(const v16h*)(bptr + ks);    // global_load x2 (b128)
      acc = __builtin_amdgcn_wmma_f32_16x16x32_f16(
          /*neg_a=*/false, a, /*neg_b=*/false, b,
          /*c_mod=*/(short)0, acc, /*reuse_a=*/false, /*reuse_b=*/false);
    }

    // C layout: lanes 0-15 hold N=lane (M = vgpr 0..7), lanes 16-31 M = 8+vgpr.
    const int n_l   = wave * 16 + (lane & 15);
    const int mbase = (lane >> 4) * 8;
#pragma unroll
    for (int v = 0; v < 8; ++v) tile[(mbase + v) * 64 + n_l] = acc[v];
    __syncthreads();

    // 16 threads (one per row) merge 64 candidates into sorted top-21 lists.
    // Strict '>' keeps earlier (lower) indices ahead on ties -> stable top_k.
    if (tid < 16) {
      float*       tvr = tv + tid * KSEL;
      int*         tir = ti + tid * KSEL;
      const float* tr  = tile + tid * 64;
      for (int j = 0; j < 64; ++j) {
        const float val = tr[j];
        if (val > tvr[KSEL - 1]) {
          int k = KSEL - 1;
          while (k > 0 && val > tvr[k - 1]) {
            tvr[k] = tvr[k - 1]; tir[k] = tir[k - 1]; --k;
          }
          tvr[k] = val; tir[k] = nc + j;
        }
      }
    }
    __syncthreads();   // protect tile before next chunk overwrites it
  }

  if (tid < 16 * KSEL) {
    const int r = tid / KSEL, k = tid % KSEL;
    topk_idx[(size_t)(row0 + r) * IDX_PITCH + k] = ti[r * KSEL + k];
  }
}

// ---------------- Kernel 3: zero the 256 MB output -------------------------
__global__ void zero_out(float4* __restrict__ out, long n4) {
  long i = (long)blockIdx.x * blockDim.x + threadIdx.x;
  const long stride = (long)gridDim.x * blockDim.x;
  const float4 z = {0.f, 0.f, 0.f, 0.f};
  for (; i < n4; i += stride) out[i] = z;
}

// ---------------- Kernel 4: scatter normalized edge weights ----------------
// top-k indices are distinct per row, so row_sum == sum(ew) + 1e-8 for all
// rows (matches the reference's scatter-then-normalize exactly).
__global__ void scatter_w(const int* __restrict__ topk_idx,
                          const float* __restrict__ ew,
                          float* __restrict__ out) {
  const int t = blockIdx.x * blockDim.x + threadIdx.x;  // row*KSEL + rank
  const int row = t / KSEL;
  const int k   = t % KSEL;
  if (row >= BN) return;
  float denom = 1e-8f;
#pragma unroll
  for (int i = 0; i < KSEL; ++i) denom += ew[i];
  const int col = topk_idx[(size_t)row * IDX_PITCH + k];
  out[(size_t)row * BN + col] = ew[k] / denom;
}

// ---------------------------------------------------------------------------
extern "C" void kernel_launch(void* const* d_in, const int* in_sizes, int n_in,
                              void* d_out, int out_size, void* d_ws, size_t ws_size,
                              hipStream_t stream) {
  const float* x  = (const float*)d_in[0];   // [8192, 1024] f32
  const float* ew = (const float*)d_in[1];   // [21] f32
  float* out = (float*)d_out;                // [8192, 8192] f32

  const size_t xn_bytes  = (size_t)BN * DK * sizeof(_Float16);      // 16 MB
  const size_t idx_bytes = (size_t)BN * IDX_PITCH * sizeof(int);    //  1 MB
  const size_t out_bytes = (size_t)out_size * sizeof(float);        // 256 MB

  // Scratch placement: prefer d_ws; fall back to carving the tail of d_out
  // for xn (xn is dead before zero_out runs, so the wipe is harmless).
  char *xn_ptr, *idx_ptr;
  if (ws_size >= xn_bytes + idx_bytes) {
    xn_ptr  = (char*)d_ws;
    idx_ptr = (char*)d_ws + xn_bytes;
  } else if (ws_size >= idx_bytes) {
    idx_ptr = (char*)d_ws;
    xn_ptr  = (char*)d_out + out_bytes - xn_bytes;
  } else {
    xn_ptr  = (char*)d_out + out_bytes - xn_bytes;
    idx_ptr = (char*)d_out + out_bytes - xn_bytes - idx_bytes;
  }
  _Float16* xn   = (_Float16*)xn_ptr;
  int*      topk = (int*)idx_ptr;

  normalize_rows<<<BN, 256, 0, stream>>>(x, xn);
  gemm_topk<<<BN / 16, 128, 0, stream>>>(xn, topk);
  zero_out<<<4096, 256, 0, stream>>>((float4*)out, (long)out_size / 4);
  scatter_w<<<(BN * KSEL + 255) / 256, 256, 0, stream>>>(topk, ew, out);
}
